// QuantumAttention64_12524124635728
// MI455X (gfx1250) — matrix operand order repaired and checked by reading, not compiled, not run on hardware
//
#include <hip/hip_runtime.h>
#include <hip/hip_bf16.h>
#include <stdint.h>

// ---------------------------------------------------------------------------
// Quantum RBF attention for MI455X (gfx1250, wave32).
//  Kernel 1: one token per wave32; 64-amp complex state held as 2 complex/lane.
//  Kernel 2: flash-style RBF attention; QK via wmma_f32_16x16x4_f32 (K=4),
//            alpha@V via wmma_f32_16x16x32_f16, V staged to LDS via
//            global_load_async_to_lds_b64 (inline asm, ASYNCcnt tracked).
// ---------------------------------------------------------------------------

typedef _Float16 v16h __attribute__((ext_vector_type(16)));
typedef float    v8f  __attribute__((ext_vector_type(8)));
typedef float    v2f  __attribute__((ext_vector_type(2)));

#define NQ   6
#define BB   16
#define SS   2048
#define DD   64

#if defined(__has_builtin)
#  if __has_builtin(__builtin_amdgcn_wmma_f32_16x16x4_f32)
#    define HAVE_WMMA_F32X4 1
#  endif
#  if __has_builtin(__builtin_amdgcn_s_wait_asynccnt)
#    define HAVE_WAIT_ASYNC_BUILTIN 1
#  endif
#endif

__device__ __forceinline__ float sx(float v, int m) { return __shfl_xor(v, m, 32); }

struct Cx { float re, im; };
__device__ __forceinline__ Cx cmul(Cx a, Cx b) { return { a.re*b.re - a.im*b.im, a.re*b.im + a.im*b.re }; }
__device__ __forceinline__ Cx cadd(Cx a, Cx b) { return { a.re + b.re, a.im + b.im }; }
__device__ __forceinline__ Cx sxC(Cx v, int m) { return { sx(v.re, m), sx(v.im, m) }; }

struct St { Cx a0, a1; };   // amplitudes: index = lane (a0), 32+lane (a1)

__device__ __forceinline__ float wred(float v) {
    for (int m = 16; m; m >>= 1) v += sx(v, m);
    return v;
}

// One-qubit gate on wire w (bit weight 1<<(5-w); wire 0 = register pair).
__device__ __forceinline__ void apply1q(St& s, int wire, Cx u00, Cx u01, Cx u10, Cx u11, int lane) {
    if (wire == 0) {
        Cx n0 = cadd(cmul(u00, s.a0), cmul(u01, s.a1));
        Cx n1 = cadd(cmul(u10, s.a0), cmul(u11, s.a1));
        s.a0 = n0; s.a1 = n1;
    } else {
        int m = 1 << (5 - wire);
        bool hi = (lane & m) != 0;
        Cx p0 = sxC(s.a0, m), p1 = sxC(s.a1, m);
        Cx ca = hi ? u10 : u00;
        Cx cb = hi ? u11 : u01;
        Cx aa0 = hi ? p0 : s.a0,  bb0 = hi ? s.a0 : p0;
        Cx aa1 = hi ? p1 : s.a1,  bb1 = hi ? s.a1 : p1;
        s.a0 = cadd(cmul(ca, aa0), cmul(cb, bb0));
        s.a1 = cadd(cmul(ca, aa1), cmul(cb, bb1));
    }
}

__device__ __forceinline__ void cnot_chain(St& s, int lane) {
    // CNOT(0,1): control = bit5 (register index) -> permute only a1 by target mask 16
    s.a1 = sxC(s.a1, 16);
    // CNOT(c,c+1) for c=1..4: both bits are lane bits
    for (int c = 1; c <= 4; ++c) {
        int mc = 1 << (5 - c), mt = mc >> 1;
        Cx p0 = sxC(s.a0, mt), p1 = sxC(s.a1, mt);
        bool ctl = (lane & mc) != 0;
        s.a0 = ctl ? p0 : s.a0;
        s.a1 = ctl ? p1 : s.a1;
    }
}

__device__ __forceinline__ void apply_pqc(St& s, const float* w, int lane) {
    for (int l = 0; l < 2; ++l) {
        for (int i = 0; i < NQ; ++i) {
            float t0 = w[(l*NQ + i)*3 + 0], t1 = w[(l*NQ + i)*3 + 1];
            float s0v, c0, s1v, c1;
            __sincosf(0.5f*t0, &s0v, &c0);
            __sincosf(0.5f*t1, &s1v, &c1);
            Cx u00 = {  c1*c0,   s1v*s0v };
            Cx u01 = { -s1v*c0, -c1*s0v  };
            Cx u10 = {  s1v*c0, -c1*s0v  };
            Cx u11 = {  c1*c0,  -s1v*s0v };
            apply1q(s, i, u00, u01, u10, u11, lane);
        }
        cnot_chain(s, lane);
        for (int i = 0; i < NQ; ++i) {
            float t2 = w[(l*NQ + i)*3 + 2];
            float s2v, c2;
            __sincosf(0.5f*t2, &s2v, &c2);
            apply1q(s, i, {c2,0.f}, {-s2v,0.f}, {s2v,0.f}, {c2,0.f}, lane);
        }
    }
}

__device__ __forceinline__ void apply_u3_all(St& s, const float* mw, int lane) {
    for (int i = 0; i < NQ; ++i) {
        float t = mw[i*3+0], p = mw[i*3+1], lam = mw[i*3+2];
        float st, ct, sp, cp, sl, cl;
        __sincosf(0.5f*t, &st, &ct);
        __sincosf(p,   &sp, &cp);
        __sincosf(lam, &sl, &cl);
        Cx u00 = { ct, 0.f };
        Cx u01 = { -cl*st, -sl*st };
        Cx u10 = {  cp*st,  sp*st };
        Cx u11 = { (cp*cl - sp*sl)*ct, (sp*cl + cp*sl)*ct };
        apply1q(s, i, u00, u01, u10, u11, lane);
    }
}

__device__ __forceinline__ void branch_qk(const St& common, const float* pqc_w, const float* u3_w,
                                          const float* pw, const float* pb,
                                          const float* g, const float* bb,
                                          float* outp, int tok, int lane) {
    St s = common;
    apply_pqc(s, pqc_w, lane);
    apply_u3_all(s, u3_w, lane);
    float p0 = s.a0.re*s.a0.re + s.a0.im*s.a0.im;
    float p1 = s.a1.re*s.a1.re + s.a1.im*s.a1.im;
    float z[NQ];
    z[0] = wred(p0 - p1);                          // wire 0: bit 5 = register index
    for (int i = 1; i < NQ; ++i) {
        int bp = 5 - i;
        float sgn = ((lane >> bp) & 1) ? -1.f : 1.f;
        z[i] = wred(sgn * (p0 + p1));
    }
    float y[4]; float mean = 0.f;
    for (int j = 0; j < 4; ++j) {
        float acc = pb[j];
        for (int i = 0; i < NQ; ++i) acc += pw[j*NQ + i] * z[i];
        y[j] = acc; mean += acc;
    }
    mean *= 0.25f;
    float var = 0.f;
    for (int j = 0; j < 4; ++j) { float d = y[j] - mean; var += d*d; }
    var *= 0.25f;
    float iv = rsqrtf(var + 1e-5f);
    if (lane == 0) {
        for (int j = 0; j < 4; ++j)
            outp[(size_t)tok*4 + j] = (y[j] - mean)*iv*g[j] + bb[j];
    }
}

// --------------------------- Kernel 1: statevector sim ----------------------
__global__ __launch_bounds__(256)
void qsim_kernel(const float* __restrict__ x, const float* __restrict__ inp_scale,
                 const float* __restrict__ enc_w, const float* __restrict__ q_w,
                 const float* __restrict__ k_w,   const float* __restrict__ v_w,
                 const float* __restrict__ mq,    const float* __restrict__ mk,
                 const float* __restrict__ mv,
                 const float* __restrict__ qpw, const float* __restrict__ qpb,
                 const float* __restrict__ kpw, const float* __restrict__ kpb,
                 const float* __restrict__ lng, const float* __restrict__ lnb,
                 float* __restrict__ qout, float* __restrict__ kout,
                 _Float16* __restrict__ vt /* [B][64][S] */) {
    int lane = threadIdx.x & 31;
    int tok  = (blockIdx.x * blockDim.x + threadIdx.x) >> 5;

    float x0 = x[(size_t)tok*DD + lane]      * inp_scale[lane];
    float x1 = x[(size_t)tok*DD + 32 + lane] * inp_scale[32 + lane];
    float nrm = sqrtf(wred(x0*x0 + x1*x1));
    float inv = 1.f / (nrm + 1e-9f);

    St s;
    s.a0 = { x0*inv, 0.f };
    s.a1 = { x1*inv, 0.f };
    apply_pqc(s, enc_w, lane);
    St common = s;

    branch_qk(common, q_w, mq, qpw, qpb, lng, lnb, qout, tok, lane);
    branch_qk(common, k_w, mk, kpw, kpb, lng, lnb, kout, tok, lane);

    // V branch: probabilities -> layernorm (no affine) -> f16, transposed layout
    s = common;
    apply_pqc(s, v_w, lane);
    apply_u3_all(s, mv, lane);
    float p0 = s.a0.re*s.a0.re + s.a0.im*s.a0.im;
    float p1 = s.a1.re*s.a1.re + s.a1.im*s.a1.im;
    float mean = wred(p0 + p1) * (1.f/64.f);
    float d0 = p0 - mean, d1 = p1 - mean;
    float var = wred(d0*d0 + d1*d1) * (1.f/64.f);
    float iv = rsqrtf(var + 1e-5f);
    int b = tok / SS, sidx = tok - b * SS;
    vt[(size_t)(b*DD + lane)*SS + sidx]        = (_Float16)(d0 * iv);
    vt[(size_t)(b*DD + 32 + lane)*SS + sidx]   = (_Float16)(d1 * iv);
}

// --------------------------- Kernel 2: RBF attention ------------------------
// Block: 256 threads = 8 waves; wave w owns 16 query rows; 64 key chunks of 32.
__global__ __launch_bounds__(256)
void attn_kernel(const float* __restrict__ qf, const float* __restrict__ kf,
                 const _Float16* __restrict__ vt, const float* __restrict__ raw_tau,
                 float* __restrict__ out) {
    const int tid = threadIdx.x, lane = tid & 31, w = tid >> 5;
    const int blocksPerB = SS / 128;
    const int b = blockIdx.x / blocksPerB;
    const int qbase = (blockIdx.x % blocksPerB) * 128 + w * 16;

    __shared__ __align__(32) _Float16 ldsV[DD * 32];   // [dim][key-in-chunk]
    __shared__ __align__(32) _Float16 ldsA[8][16 * 32]; // per-wave alpha staging

    const float tau = log1pf(__expf(raw_tau[0])) + 1e-9f;
    const float inv_tau = 1.f / tau;

    // ---- Q fragment: A-matrix 16x4 f32 (lanes 0-15: M=lane,K=0..1; 16-31: K=2..3)
    const int mrow = lane & 15;
    const int kb   = (lane >> 4) * 2;
    const int add  = (lane >= 16) ? 8 : 0;
    v2f aq;
    {
        const float* qp = qf + (size_t)(b*SS + qbase + mrow)*4 + kb;
        aq.x = qp[0]; aq.y = qp[1];
    }
    float q2full = aq.x*aq.x + aq.y*aq.y;
    q2full += sx(q2full, 16);                       // |q|^2 of row (lane&15)
    float q2c[8];
    for (int r = 0; r < 8; ++r) q2c[r] = __shfl(q2full, r + add, 32);

#ifndef HAVE_WMMA_F32X4
    float qv[4];
    {
        float ox = sx(aq.x, 16), oy = sx(aq.y, 16);
        if (lane < 16) { qv[0]=aq.x; qv[1]=aq.y; qv[2]=ox; qv[3]=oy; }
        else           { qv[0]=ox;   qv[1]=oy;   qv[2]=aq.x; qv[3]=aq.y; }
    }
#endif

    v8f acc[4];
    for (int n = 0; n < 4; ++n) acc[n] = (v8f){0,0,0,0,0,0,0,0};
    float rs[8];
    for (int r = 0; r < 8; ++r) rs[r] = 0.f;

    for (int kc = 0; kc < SS; kc += 32) {
        // ---- stage V chunk (64 dims x 32 keys, f16) into LDS via async DMA.
        // ASYNCcnt-tracked; LDS byte offset = low 32 bits of flat shared addr.
        for (int it = 0; it < 2; ++it) {
            int lin  = tid + it * 256;              // 0..511, 8 bytes each
            int row  = lin >> 3;                    // dim 0..63
            int off4 = (lin & 7) * 4;               // half offset within 32
            const _Float16* gp = vt + (size_t)(b*DD + row)*SS + kc + off4;
            uint64_t ga = (uint64_t)(uintptr_t)gp;
            uint32_t la = (uint32_t)(uintptr_t)&ldsV[row*32 + off4];
            asm volatile("global_load_async_to_lds_b64 %0, %1, off"
                         :: "v"(la), "v"(ga) : "memory");
        }
#ifdef HAVE_WAIT_ASYNC_BUILTIN
        __builtin_amdgcn_s_wait_asynccnt(0);
#else
        asm volatile("s_wait_asynccnt 0" ::: "memory");
#endif
        __syncthreads();

        // ---- scores: two 16x16 tiles (j = key sub-tile)
        for (int j = 0; j < 2; ++j) {
            v2f bk;
            {
                const float* kp = kf + (size_t)(b*SS + kc + j*16 + mrow)*4 + kb;
                bk.x = kp[0]; bk.y = kp[1];
            }
            float k2c = bk.x*bk.x + bk.y*bk.y;
            k2c += sx(k2c, 16);                     // |k|^2 of column (lane&15)

            v8f sc;
#ifdef HAVE_WMMA_F32X4
            {
                v8f cz = (v8f){0,0,0,0,0,0,0,0};
                sc = __builtin_amdgcn_wmma_f32_16x16x4_f32(
                         false, aq, false, bk, (short)0, cz, false, false);
            }
#else
            {
                float kv[4];
                float ox = sx(bk.x, 16), oy = sx(bk.y, 16);
                if (lane < 16) { kv[0]=bk.x; kv[1]=bk.y; kv[2]=ox; kv[3]=oy; }
                else           { kv[0]=ox;   kv[1]=oy;   kv[2]=bk.x; kv[3]=bk.y; }
                for (int r = 0; r < 8; ++r) {
                    float d = 0.f;
                    for (int c4 = 0; c4 < 4; ++c4)
                        d += __shfl(qv[c4], r + add, 32) * kv[c4];
                    sc[r] = d;
                }
            }
#endif
            for (int r = 0; r < 8; ++r) {
                float d2 = q2c[r] + k2c - 2.f * sc[r];
                d2 = fmaxf(d2, 0.f);
                float alpha = __expf(-d2 * inv_tau);
                rs[r] += alpha;
                int M = r + add;                     // C-layout row
                ldsA[w][M*32 + j*16 + mrow] = (_Float16)alpha;
            }
        }
        asm volatile("s_wait_dscnt 0" ::: "memory"); // wave-local staging visible

        // ---- gather alpha as f16 A-fragment (16x32)
        v16h aA;
        {
            const int Mb = mrow * 32;
            for (int p = 0; p < 8; ++p) {
                int K = 2*p + 2*add;
                aA[2*p]   = ldsA[w][Mb + K];
                aA[2*p+1] = ldsA[w][Mb + K + 1];
            }
        }
        // ---- B fragments from LDS and accumulate alpha @ V
        const int kb2 = (lane >= 16) ? 16 : 0;
        for (int n = 0; n < 4; ++n) {
            const v16h bV = *(const v16h*)&ldsV[(n*16 + mrow)*32 + kb2];
            acc[n] = __builtin_amdgcn_wmma_f32_16x16x32_f16(
                         false, aA, false, bV, (short)0, acc[n], false, false);
        }
        __syncthreads();   // protect ldsV before next chunk overwrite
    }

    // ---- rowsum reduce across the 16-lane half, normalize, store
    for (int r = 0; r < 8; ++r)
        for (int m = 1; m < 16; m <<= 1) rs[r] += sx(rs[r], m);

    for (int n = 0; n < 4; ++n)
        for (int r = 0; r < 8; ++r) {
            int row = qbase + r + add;
            int col = n*16 + mrow;
            out[(size_t)(b*SS + row)*DD + col] = acc[n][r] / (rs[r] + 1e-9f);
        }
}

// ---------------------------------------------------------------------------
extern "C" void kernel_launch(void* const* d_in, const int* in_sizes, int n_in,
                              void* d_out, int out_size, void* d_ws, size_t ws_size,
                              hipStream_t stream) {
    (void)in_sizes; (void)n_in; (void)out_size; (void)ws_size;
    const float* x         = (const float*)d_in[0];
    const float* inp_scale = (const float*)d_in[1];
    const float* enc_w     = (const float*)d_in[2];
    const float* q_w       = (const float*)d_in[3];
    const float* k_w       = (const float*)d_in[4];
    const float* v_w       = (const float*)d_in[5];
    const float* mq        = (const float*)d_in[6];
    const float* mk        = (const float*)d_in[7];
    const float* mv        = (const float*)d_in[8];
    const float* qpw       = (const float*)d_in[9];
    const float* qpb       = (const float*)d_in[10];
    const float* kpw       = (const float*)d_in[11];
    const float* kpb       = (const float*)d_in[12];
    const float* lng       = (const float*)d_in[13];
    const float* lnb       = (const float*)d_in[14];
    const float* raw_tau   = (const float*)d_in[15];

    const int bsz = BB * SS;
    float*    qf = (float*)d_ws;                    // bsz*4 floats
    float*    kf = qf + (size_t)bsz * 4;            // bsz*4 floats
    _Float16* vt = (_Float16*)(kf + (size_t)bsz*4); // B*64*S halves

    qsim_kernel<<<bsz / 8, 256, 0, stream>>>(x, inp_scale, enc_w, q_w, k_w, v_w,
                                             mq, mk, mv, qpw, qpb, kpw, kpb,
                                             lng, lnb, qf, kf, vt);
    attn_kernel<<<BB * (SS / 128), 256, 0, stream>>>(qf, kf, vt, raw_tau,
                                                     (float*)d_out);
}
